// SinkhornAttention_21328807592570
// MI455X (gfx1250) — compile-verified, matrix operand order checked by reading, not run on hardware
//
#include <hip/hip_runtime.h>
#include <hip/hip_bf16.h>
#include <cstdint>

typedef __attribute__((ext_vector_type(16))) _Float16 v16h;
typedef __attribute__((ext_vector_type(8)))  float    v8f;
typedef __attribute__((ext_vector_type(4)))  unsigned int v4u;
typedef __attribute__((ext_vector_type(8)))  unsigned int v8u;

#define BH      32
#define BUCKETS 64
#define BSZ     128
#define DH      64
#define KEYS    256           // 2*BSZ
#define KSTRIDE 72            // ldsK row stride in halves (16B aligned)
#define VSTRIDE 272           // ldsVT row stride in halves (16B aligned)
#define TILE_ELEMS (BSZ * DH) // 8192 f32 per bucket tile (32KB, contiguous)

// shared-memory arena layout (bytes); single __shared__ var => LDS offset 0
#define OFF_K   0u
#define OFF_VT  (OFF_K  + KEYS * KSTRIDE * 2u)   // 36864
#define OFF_P   (OFF_VT + DH * VSTRIDE * 2u)     // +34816 -> 71680
#define OFF_RAW (OFF_P  + 8u * 16u * 32u * 2u)   // +8192  -> 79872
#define SMEM_BYTES (OFF_RAW + 2u * TILE_ELEMS * 4u)  // +65536 -> 145408

// ---------------------------------------------------------------------------
// Kernel A: SortNet. Per bh: bucket sums -> R = relu(x @ W) -> softmax ->
// top-1 (index, weight) per bucket row.
// ---------------------------------------------------------------------------
__global__ __launch_bounds__(256)
void sortnet_kernel(const float* __restrict__ q, const float* __restrict__ k,
                    const float* __restrict__ W,
                    int* __restrict__ selIdx, float* __restrict__ selW) {
    __shared__ float xs[BUCKETS][2 * DH];   // 32 KB
    __shared__ float Wls[2 * DH * BUCKETS]; // 32 KB

    const int tid = threadIdx.x;
    const int bhi = blockIdx.x;
    const int hi  = bhi & 7;                // bh = b*8 + h

    for (int p = tid; p < BUCKETS * 2 * DH; p += 256) {
        const int u = p >> 7;
        const int d = p & 127;
        const float* src = (d < DH)
            ? q + ((size_t)(bhi * BUCKETS + u) * BSZ) * DH + d
            : k + ((size_t)(bhi * BUCKETS + u) * BSZ) * DH + (d - DH);
        float s = 0.f;
        for (int t2 = 0; t2 < BSZ; ++t2) s += src[t2 * DH];
        xs[u][d] = s;
    }
    const float* Wg = W + (size_t)hi * 2 * DH * BUCKETS;
    for (int p = tid; p < 2 * DH * BUCKETS; p += 256) Wls[p] = Wg[p];
    __syncthreads();

    if (tid < BUCKETS) {
        const int u = tid;
        float r[BUCKETS];
        float mx = -1e30f;
        #pragma unroll 4
        for (int v = 0; v < BUCKETS; ++v) {
            float acc = 0.f;
            #pragma unroll 8
            for (int d = 0; d < 2 * DH; ++d) acc += xs[u][d] * Wls[d * BUCKETS + v];
            acc = fmaxf(acc, 0.f);          // relu
            r[v] = acc;
            mx = fmaxf(mx, acc);
        }
        float se = 0.f, best = -1e30f;
        int bi = 0;
        #pragma unroll 4
        for (int v = 0; v < BUCKETS; ++v) {
            se += __expf(r[v] - mx);
            if (r[v] > best) { best = r[v]; bi = v; }   // first-max like argmax
        }
        selIdx[bhi * BUCKETS + u] = bi;
        selW[bhi * BUCKETS + u]   = __expf(best - mx) / se;
    }
}

// ---------------------------------------------------------------------------
// TDM: 1-D tile load (contiguous `TILE_ELEMS` f32) from global to LDS offset.
// Descriptor per CDNA5 ISA 8.3/8.4: group0 = {count/flags, lds_addr,
// global_addr, type=2}; group1 = {data_size=4B, tensor_dim0/1, tile_dim0,
// tensor_dim0_stride}. Issued once per wave (EXEC ignored by TDM).
// ---------------------------------------------------------------------------
__device__ inline void tdm_load_tile(unsigned lds_off, const void* gptr) {
    const unsigned long long ga = (unsigned long long)(uintptr_t)gptr;
    v4u g0;
    g0[0] = 1u;                                            // count=1, user desc
    g0[1] = lds_off;                                       // lds_addr (bytes)
    g0[2] = (unsigned)(ga & 0xffffffffu);                  // global_addr[31:0]
    g0[3] = (unsigned)((ga >> 32) & 0x01ffffffu)           // global_addr[56:32]
          | (2u << 30);                                    // type = 2 (image)
    v8u g1;
    g1[0] = (2u << 16);                                    // data_size = 4B
    g1[1] = (TILE_ELEMS & 0xffffu) << 16;                  // tensor_dim0[15:0]
    g1[2] = ((TILE_ELEMS >> 16) & 0xffffu) | (1u << 16);   // dim0 hi | tensor_dim1=1
    g1[3] = (TILE_ELEMS & 0xffffu) << 16;                  // tile_dim0 = 8192
    g1[4] = 0u;                                            // tile_dim1/2 unused
    g1[5] = (unsigned)TILE_ELEMS;                          // tensor_dim0_stride lo
    g1[6] = 0u;
    g1[7] = 0u;
    asm volatile("tensor_load_to_lds %0, %1" :: "s"(g0), "s"(g1) : "memory");
}

// ---------------------------------------------------------------------------
// Fragment loaders (ISA 7.12.2 16-bit A/B layouts, wave32)
// ---------------------------------------------------------------------------
__device__ inline v16h load_frag_lds(const _Float16* rowChunk, int lane) {
    const int boff = (lane < 16) ? 0 : 8;
    union { v16h h; uint4 u[2]; } t;
    t.u[0] = *(const uint4*)(rowChunk + boff);       // K = base .. base+7
    t.u[1] = *(const uint4*)(rowChunk + boff + 16);  // K = base+16 .. base+23
    return t.h;
}

__device__ inline v16h load_qfrag(const float* rowChunk, int lane) {
    const int boff = (lane < 16) ? 0 : 8;
    const float* p = rowChunk + boff;
    v16h h;
    #pragma unroll
    for (int i = 0; i < 8; ++i) h[i] = (_Float16)p[i];
    #pragma unroll
    for (int i = 0; i < 8; ++i) h[8 + i] = (_Float16)p[16 + i];
    return h;
}

// ---------------------------------------------------------------------------
// Kernel B: bucketed flash attention over [w*k[v*] ; k[u]] with f16 WMMA,
// K/V tiles staged through the Tensor Data Mover.
// ---------------------------------------------------------------------------
__global__ __launch_bounds__(256)
void bucket_attn_kernel(const float* __restrict__ q, const float* __restrict__ k,
                        const float* __restrict__ v,
                        const int* __restrict__ selIdx, const float* __restrict__ selW,
                        float* __restrict__ out) {
    __shared__ __attribute__((aligned(16))) unsigned char smem[SMEM_BYTES];
    _Float16* ldsK  = (_Float16*)(smem + OFF_K);    // key-major (B of Q.K^T)
    _Float16* ldsVT = (_Float16*)(smem + OFF_VT);   // dim-major (B of P.V)
    _Float16* ldsP  = (_Float16*)(smem + OFF_P);    // per-wave P relayout
    float*    raw   = (float*)(smem + OFF_RAW);     // TDM f32 staging (2 tiles)

    const int tid  = threadIdx.x;
    const int lane = tid & 31;
    const int wv   = tid >> 5;
    const int u    = blockIdx.x & (BUCKETS - 1);
    const int bhi  = blockIdx.x >> 6;
    const int bb   = bhi * BUCKETS;

    const int   vsel = selIdx[bb + u];
    const float wsel = selW[bb + u];

    const float* ksel = k + (size_t)(bb + vsel) * TILE_ELEMS;
    const float* kown = k + (size_t)(bb + u)    * TILE_ELEMS;
    const float* vsl  = v + (size_t)(bb + vsel) * TILE_ELEMS;
    const float* vown = v + (size_t)(bb + u)    * TILE_ELEMS;

    // Q fragments for this wave's 16 rows (global only; independent of LDS).
    const int rowbase = wv * 16;
    const float* qrow = q + ((size_t)(bb + u) * BSZ + rowbase + (lane & 15)) * DH;
    const v16h aq0 = load_qfrag(qrow + 0,  lane);
    const v16h aq1 = load_qfrag(qrow + 32, lane);

    // ---- Phase A: TDM-load both K tiles, convert f32->f16 (scale top half) ----
    if (wv == 0) {
        tdm_load_tile(OFF_RAW,                    ksel);
        tdm_load_tile(OFF_RAW + TILE_ELEMS * 4u,  kown);
        __builtin_amdgcn_s_wait_tensorcnt(0);
    }
    __syncthreads();
    for (int idx = tid; idx < KEYS * DH; idx += 256) {
        const int row = idx >> 6;
        const int d   = idx & 63;
        float f = raw[idx];                       // tiles are contiguous: rows 0..255
        if (row < BSZ) f *= wsel;
        ldsK[row * KSTRIDE + d] = (_Float16)f;
    }
    __syncthreads();   // raw fully consumed before reuse

    // ---- Phase B: TDM-load both V tiles, convert to transposed f16 layout ----
    if (wv == 0) {
        tdm_load_tile(OFF_RAW,                    vsl);
        tdm_load_tile(OFF_RAW + TILE_ELEMS * 4u,  vown);
        __builtin_amdgcn_s_wait_tensorcnt(0);
    }
    __syncthreads();
    for (int idx = tid; idx < KEYS * DH; idx += 256) {
        const int row = idx >> 6;
        const int d   = idx & 63;
        float f = raw[idx];
        if (row < BSZ) f *= wsel;
        ldsVT[d * VSTRIDE + row] = (_Float16)f;
    }
    __syncthreads();

    // ---- Flash attention main loop ----
    v8f o0 = {}, o1 = {}, o2 = {}, o3 = {};
    float m[8], lsum[8];
    #pragma unroll
    for (int r = 0; r < 8; ++r) { m[r] = -1e30f; lsum[r] = 0.f; }

    const float scale = 0.044194173824159216f;   // DIM(512)^-0.5
    const int   n     = lane & 15;
    const int   mbase = (lane < 16) ? 0 : 8;
    _Float16*   pbuf  = ldsP + wv * (16 * 32);

    for (int kb = 0; kb < KEYS; kb += 32) {
        const _Float16* rA = &ldsK[(kb + n)      * KSTRIDE];
        const _Float16* rB = &ldsK[(kb + 16 + n) * KSTRIDE];
        const v16h b00 = load_frag_lds(rA + 0,  lane);
        const v16h b01 = load_frag_lds(rA + 32, lane);
        const v16h b10 = load_frag_lds(rB + 0,  lane);
        const v16h b11 = load_frag_lds(rB + 32, lane);

        const v8f zero = {};
        v8f s0 = __builtin_amdgcn_wmma_f32_16x16x32_f16(false, aq0, false, b00, (short)0, zero, false, false);
        s0     = __builtin_amdgcn_wmma_f32_16x16x32_f16(false, aq1, false, b01, (short)0, s0,   false, false);
        v8f s1 = __builtin_amdgcn_wmma_f32_16x16x32_f16(false, aq0, false, b10, (short)0, zero, false, false);
        s1     = __builtin_amdgcn_wmma_f32_16x16x32_f16(false, aq1, false, b11, (short)0, s1,   false, false);

        #pragma unroll
        for (int r = 0; r < 8; ++r) {
            const float a0 = s0[r] * scale;
            const float a1 = s1[r] * scale;
            float rm = fmaxf(a0, a1);
            rm = fmaxf(rm, __shfl_xor(rm, 1));
            rm = fmaxf(rm, __shfl_xor(rm, 2));
            rm = fmaxf(rm, __shfl_xor(rm, 4));
            rm = fmaxf(rm, __shfl_xor(rm, 8));
            const float mnew  = fmaxf(m[r], rm);
            const float alpha = __expf(m[r] - mnew);
            const float p0 = __expf(a0 - mnew);
            const float p1 = __expf(a1 - mnew);
            float rs = p0 + p1;
            rs += __shfl_xor(rs, 1);
            rs += __shfl_xor(rs, 2);
            rs += __shfl_xor(rs, 4);
            rs += __shfl_xor(rs, 8);
            lsum[r] = lsum[r] * alpha + rs;
            m[r] = mnew;
            o0[r] *= alpha; o1[r] *= alpha; o2[r] *= alpha; o3[r] *= alpha;
            pbuf[(mbase + r) * 32 + n]      = (_Float16)p0;
            pbuf[(mbase + r) * 32 + 16 + n] = (_Float16)p1;
        }

        const v16h ap  = load_frag_lds(pbuf + (lane & 15) * 32, lane);
        const v16h bv0 = load_frag_lds(&ldsVT[(n)      * VSTRIDE + kb], lane);
        const v16h bv1 = load_frag_lds(&ldsVT[(16 + n) * VSTRIDE + kb], lane);
        const v16h bv2 = load_frag_lds(&ldsVT[(32 + n) * VSTRIDE + kb], lane);
        const v16h bv3 = load_frag_lds(&ldsVT[(48 + n) * VSTRIDE + kb], lane);
        o0 = __builtin_amdgcn_wmma_f32_16x16x32_f16(false, ap, false, bv0, (short)0, o0, false, false);
        o1 = __builtin_amdgcn_wmma_f32_16x16x32_f16(false, ap, false, bv1, (short)0, o1, false, false);
        o2 = __builtin_amdgcn_wmma_f32_16x16x32_f16(false, ap, false, bv2, (short)0, o2, false, false);
        o3 = __builtin_amdgcn_wmma_f32_16x16x32_f16(false, ap, false, bv3, (short)0, o3, false, false);
    }

    // epilogue: normalize and store (f32 out)
    float* op = out + ((size_t)(bb + u) * BSZ + rowbase) * DH;
    #pragma unroll
    for (int r = 0; r < 8; ++r) {
        const float inv = 1.0f / lsum[r];
        const int rowoff = (mbase + r) * DH;
        op[rowoff + 0  + n] = o0[r] * inv;
        op[rowoff + 16 + n] = o1[r] * inv;
        op[rowoff + 32 + n] = o2[r] * inv;
        op[rowoff + 48 + n] = o3[r] * inv;
    }
}

// ---------------------------------------------------------------------------
extern "C" void kernel_launch(void* const* d_in, const int* in_sizes, int n_in,
                              void* d_out, int out_size, void* d_ws, size_t ws_size,
                              hipStream_t stream) {
    (void)in_sizes; (void)n_in; (void)out_size; (void)ws_size;
    const float* q = (const float*)d_in[0];
    const float* k = (const float*)d_in[1];
    const float* v = (const float*)d_in[2];
    const float* W = (const float*)d_in[3];
    float* out = (float*)d_out;

    int*   selIdx = (int*)d_ws;
    float* selW   = (float*)((char*)d_ws + BH * BUCKETS * sizeof(int));

    sortnet_kernel<<<BH, 256, 0, stream>>>(q, k, W, selIdx, selW);
    bucket_attn_kernel<<<BH * BUCKETS, 256, 0, stream>>>(q, k, v, selIdx, selW, out);
}
